// CGRUCell_52235392254569
// MI455X (gfx1250) — compile-verified, matrix operand order checked by reading, not run on hardware
//
#include <hip/hip_runtime.h>
#include <hip/hip_bf16.h>
#include <math.h>

// ---------------------------------------------------------------------------
// CDNA5 (gfx1250) fused Bahdanau-attention GRU cell.
// All heavy GEMMs run on v_wmma_f32_16x16x32_f16 (f16 in, f32 accumulate).
// Wk (reused by all 512 attention workgroups) is pre-converted to f16 so the
// dominant K-loop loads B fragments with 2x global_load_b128 and no cvt VALU.
// ---------------------------------------------------------------------------

typedef __attribute__((ext_vector_type(16))) _Float16 v16h;
typedef __attribute__((ext_vector_type(8)))  _Float16 v8h;
typedef __attribute__((ext_vector_type(8)))  float    v8f;

#define B_DIM   64
#define S_DIM   1024
#define C_DIM   1024
#define A_DIM   1024
#define H_DIM   1024
#define STILE   128
#define LDH     1032   // LDS halfs per row: 2064B stride -> conflict-free b128 frag loads

__device__ __forceinline__ v8f wmma_f16(v16h a, v16h b, v8f c) {
    // D = A(16x32 f16) x B(32x16 f16) + C(16x16 f32)
    return __builtin_amdgcn_wmma_f32_16x16x32_f16(
        /*neg_a=*/false, a, /*neg_b=*/false, b,
        /*c_mod=*/(short)0, c, /*reuse_a=*/false, /*reuse_b=*/false);
}

// Load one 16x32 f16 operand fragment from a row-major f32 matrix.
// Lane mapping (ISA 7.12.2, 16-bit A 16x32): row = lane&15,
// elems 0..7 -> K = (lane>>4)*8 + i ; elems 8..15 -> K = 16 + (lane>>4)*8 + i.
__device__ __forceinline__ v16h frag_from_f32(const float* __restrict__ base,
                                              int ld, int lane) {
    const float* p = base + (size_t)(lane & 15) * ld + ((lane >> 4) << 3);
    v16h r;
#pragma unroll
    for (int i = 0; i < 8; ++i) r[i] = (_Float16)p[i];
#pragma unroll
    for (int i = 0; i < 8; ++i) r[i + 8] = (_Float16)p[16 + i];
    return r;
}

// Same fragment pattern from an f16 row-major matrix (global or LDS):
// exactly 2 x b128 loads, no conversion.
__device__ __forceinline__ v16h frag_from_f16(const _Float16* __restrict__ base,
                                              int ld, int lane) {
    const _Float16* p = base + (size_t)(lane & 15) * ld + ((lane >> 4) << 3);
    v8h lo = *(const v8h*)p;
    v8h hi = *(const v8h*)(p + 16);
    v16h r;
#pragma unroll
    for (int i = 0; i < 8; ++i) { r[i] = lo[i]; r[i + 8] = hi[i]; }
    return r;
}

// f32 -> f16 bulk convert (float4 in, 4 halfs out per thread-iteration)
__global__ void cvt_f16_kernel(const float* __restrict__ src,
                               _Float16* __restrict__ dst, int n4) {
    const int i = blockIdx.x * blockDim.x + threadIdx.x;
    if (i >= n4) return;
    const float4 v = ((const float4*)src)[i];
    _Float16* d = dst + ((size_t)i << 2);
    d[0] = (_Float16)v.x; d[1] = (_Float16)v.y;
    d[2] = (_Float16)v.z; d[3] = (_Float16)v.w;
}

// ---------------------------------------------------------------------------
// Generic C[M,N] = A[M,K] @ W[N,K]^T (+bias) (+ A2[M,K2] @ W2[N,K2]^T + bias2)
// One 16x16 output tile per wave.
// ---------------------------------------------------------------------------
__global__ void gemm_xwt_bias(const float* __restrict__ A,  const float* __restrict__ W,
                              const float* __restrict__ bias,
                              const float* __restrict__ A2, const float* __restrict__ W2,
                              const float* __restrict__ bias2,
                              float* __restrict__ C, int M, int N, int K, int K2) {
    const int lane   = threadIdx.x & 31;
    const int wave   = blockIdx.x * (blockDim.x >> 5) + (threadIdx.x >> 5);
    const int ntiles = N >> 4;
    if (wave >= (M >> 4) * ntiles) return;   // wave-uniform: EXEC all-1s inside
    const int m0 = (wave / ntiles) << 4;
    const int n0 = (wave % ntiles) << 4;

    v8f acc = {};
    const float* Ab = A + (size_t)m0 * K;
    const float* Wb = W + (size_t)n0 * K;
    for (int k0 = 0; k0 < K; k0 += 32) {
        __builtin_prefetch(Wb + k0 + 64, 0, 0);
        v16h af = frag_from_f32(Ab + k0, K, lane);
        v16h bf = frag_from_f32(Wb + k0, K, lane);
        acc = wmma_f16(af, bf, acc);
    }
    if (A2 != nullptr) {
        const float* A2b = A2 + (size_t)m0 * K2;
        const float* W2b = W2 + (size_t)n0 * K2;
        for (int k0 = 0; k0 < K2; k0 += 32) {
            v16h af = frag_from_f32(A2b + k0, K2, lane);
            v16h bf = frag_from_f32(W2b + k0, K2, lane);
            acc = wmma_f16(af, bf, acc);
        }
    }
    const int n = n0 + (lane & 15);
    float bs = bias  ? bias[n]  : 0.0f;
    if (bias2) bs += bias2[n];
#pragma unroll
    for (int j = 0; j < 8; ++j) {
        const int m = m0 + ((lane >> 4) << 3) + j;
        C[(size_t)m * N + n] = acc[j] + bs;
    }
}

// ---------------------------------------------------------------------------
// Fused attention logits: for one (b, 128-row S tile)
//   logits[s] = sum_a tanh( (context[s,:] . Wk[a,:]) + q[b,a] + bk[a] ) * Wl[a] + bl
// context tile staged in LDS as f16; Wk pre-converted to f16 in workspace.
// Never materializes the [B,S,A] cache (saves 536 MB of HBM round-trip).
// ---------------------------------------------------------------------------
__global__ void attn_logits_kernel(const float* __restrict__ context,    // [B,S,C] f32
                                   const _Float16* __restrict__ Wk16,    // [A,C] f16
                                   const float* __restrict__ q,          // [B,A]
                                   const float* __restrict__ bk,         // [A]
                                   const float* __restrict__ Wl,         // [A]
                                   const float* __restrict__ bl,         // [1]
                                   float* __restrict__ logits) {         // [B,S]
    extern __shared__ char smem[];
    _Float16* As  = (_Float16*)smem;                                     // [STILE][LDH] f16
    float*    qs  = (float*)(smem + (size_t)STILE * LDH * sizeof(_Float16)); // [A]
    float*    slg = qs + A_DIM;                                          // [STILE]

    const int tid  = threadIdx.x;
    const int lane = tid & 31;
    const int wv   = tid >> 5;                  // 8 waves
    const int b    = blockIdx.y;
    const int s0   = blockIdx.x * STILE;
    const float* ctx = context + ((size_t)b * S_DIM + s0) * C_DIM;

    // Stage context tile f32 -> f16 (float4 reads, packed cvt, coalesced)
    for (int idx = tid; idx < STILE * (C_DIM >> 2); idx += blockDim.x) {
        const int r  = idx >> 8;              // C_DIM/4 == 256
        const int c4 = idx & 255;
        const float4 v = ((const float4*)(ctx + (size_t)r * C_DIM))[c4];
        _Float16* d = As + (size_t)r * LDH + (c4 << 2);
        d[0] = (_Float16)v.x; d[1] = (_Float16)v.y;
        d[2] = (_Float16)v.z; d[3] = (_Float16)v.w;
    }
    for (int n = tid; n < A_DIM; n += blockDim.x) qs[n] = q[(size_t)b * A_DIM + n] + bk[n];
    for (int m = tid; m < STILE; m += blockDim.x) slg[m] = 0.0f;
    __syncthreads();

    // Each wave sweeps 8 of the 64 A-tiles (interleaved by wave id).
    for (int i = 0; i < (A_DIM >> 4) / 8; ++i) {
        const int n0 = (wv + (i << 3)) << 4;
        const _Float16* Wkb = Wk16 + (size_t)n0 * C_DIM;

        v8f acc[STILE / 16];
#pragma unroll
        for (int mt = 0; mt < STILE / 16; ++mt) acc[mt] = (v8f){};

        for (int k0 = 0; k0 < C_DIM; k0 += 32) {
            __builtin_prefetch(Wkb + k0 + 128, 0, 0);   // global_prefetch_b8
            v16h bf = frag_from_f16(Wkb + k0, C_DIM, lane);  // 2x global_load_b128
#pragma unroll
            for (int mt = 0; mt < STILE / 16; ++mt) {
                v16h af = frag_from_f16(As + (size_t)(mt << 4) * LDH + k0, LDH, lane); // ds_load_b128
                acc[mt] = wmma_f16(af, bf, acc[mt]);
            }
        }

        // Epilogue: tanh + weighted reduce over the 16 'a' columns of this tile.
        const float qv = qs[n0 + (lane & 15)];
        const float wl = Wl[n0 + (lane & 15)];
#pragma unroll
        for (int mt = 0; mt < STILE / 16; ++mt) {
#pragma unroll
            for (int j = 0; j < 8; ++j) {
                const int m = (mt << 4) + ((lane >> 4) << 3) + j;
                atomicAdd(&slg[m], tanhf(acc[mt][j] + qv) * wl);  // ds_add_f32
            }
        }
    }
    __syncthreads();

    const float bl0 = bl[0];
    for (int m = tid; m < STILE; m += blockDim.x)
        logits[(size_t)b * S_DIM + s0 + m] = slg[m] + bl0;
}

// ---------------------------------------------------------------------------
// Softmax over S per batch row, with mask (True -> -1e18).
// ---------------------------------------------------------------------------
__global__ void softmax_kernel(const float* __restrict__ logits,
                               const unsigned char* __restrict__ mask,
                               float* __restrict__ weights) {
    __shared__ float red[256];
    const int b = blockIdx.x, tid = threadIdx.x;

    float mx = -3.0e38f;
    for (int s = tid; s < S_DIM; s += 256) {
        const float v = mask[(size_t)b * S_DIM + s] ? -1.0e18f : logits[(size_t)b * S_DIM + s];
        mx = fmaxf(mx, v);
    }
    red[tid] = mx; __syncthreads();
    for (int off = 128; off > 0; off >>= 1) {
        if (tid < off) red[tid] = fmaxf(red[tid], red[tid + off]);
        __syncthreads();
    }
    mx = red[0]; __syncthreads();

    float sum = 0.0f;
    for (int s = tid; s < S_DIM; s += 256) {
        const float v = mask[(size_t)b * S_DIM + s] ? -1.0e18f : logits[(size_t)b * S_DIM + s];
        const float e = __expf(v - mx);
        weights[(size_t)b * S_DIM + s] = e;
        sum += e;
    }
    red[tid] = sum; __syncthreads();
    for (int off = 128; off > 0; off >>= 1) {
        if (tid < off) red[tid] += red[tid + off];
        __syncthreads();
    }
    const float inv = 1.0f / red[0];
    __syncthreads();
    for (int s = tid; s < S_DIM; s += 256)
        weights[(size_t)b * S_DIM + s] *= inv;
}

__global__ void zero_kernel(float* __restrict__ p, int n) {
    const int i = blockIdx.x * blockDim.x + threadIdx.x;
    if (i < n) p[i] = 0.0f;
}

// attn[b,c] = sum_s weights[b,s] * context[b,s,c]   (S split across grid.y)
__global__ void attn_wsum_kernel(const float* __restrict__ weights,
                                 const float* __restrict__ context,
                                 float* __restrict__ attn, int schunk) {
    const int b  = blockIdx.x;
    const int s0 = blockIdx.y * schunk;
    const int c4 = threadIdx.x;                    // 256 threads * float4 = 1024 cols
    const float* ctx = context + ((size_t)b * S_DIM + s0) * C_DIM;
    float4 acc = make_float4(0.f, 0.f, 0.f, 0.f);
    for (int s = 0; s < schunk; ++s) {
        const float w = weights[(size_t)b * S_DIM + s0 + s];
        const float4 v = ((const float4*)(ctx + (size_t)s * C_DIM))[c4];
        acc.x += w * v.x; acc.y += w * v.y; acc.z += w * v.z; acc.w += w * v.w;
    }
    float* dst = attn + (size_t)b * C_DIM + (c4 << 2);
    atomicAdd(dst + 0, acc.x); atomicAdd(dst + 1, acc.y);
    atomicAdd(dst + 2, acc.z); atomicAdd(dst + 3, acc.w);
}

// GRU elementwise finisher; also emits attn_values as output #2.
__global__ void gru_kernel(const float* __restrict__ gx, const float* __restrict__ gh,
                           const float* __restrict__ hidden, const float* __restrict__ attn,
                           float* __restrict__ out) {
    const int i = blockIdx.x * blockDim.x + threadIdx.x;
    if (i >= B_DIM * H_DIM) return;
    const int bb = i / H_DIM, h = i - bb * H_DIM;
    const size_t base = (size_t)bb * 3 * H_DIM;
    const float xr = gx[base + h],             hr = gh[base + h];
    const float xz = gx[base + H_DIM + h],     hz = gh[base + H_DIM + h];
    const float xn = gx[base + 2 * H_DIM + h], hn = gh[base + 2 * H_DIM + h];
    const float r = 1.0f / (1.0f + __expf(-(xr + hr)));
    const float z = 1.0f / (1.0f + __expf(-(xz + hz)));
    const float n = tanhf(xn + r * hn);
    out[i] = (1.0f - z) * n + z * hidden[i];
    out[(size_t)B_DIM * H_DIM + i] = attn[i];
}

// ---------------------------------------------------------------------------
extern "C" void kernel_launch(void* const* d_in, const int* in_sizes, int n_in,
                              void* d_out, int out_size, void* d_ws, size_t ws_size,
                              hipStream_t stream) {
    const float* input   = (const float*)d_in[0];
    const float* hidden  = (const float*)d_in[1];
    const float* context = (const float*)d_in[2];
    const unsigned char* mask = (const unsigned char*)d_in[3];
    const float* Wq  = (const float*)d_in[4];
    const float* bq  = (const float*)d_in[5];
    const float* Wk  = (const float*)d_in[6];
    const float* bk  = (const float*)d_in[7];
    const float* Wl  = (const float*)d_in[8];
    const float* bl  = (const float*)d_in[9];
    const float* We  = (const float*)d_in[10];
    const float* be  = (const float*)d_in[11];
    const float* Wa  = (const float*)d_in[12];
    const float* ba  = (const float*)d_in[13];
    const float* Wih = (const float*)d_in[14];
    const float* Whh = (const float*)d_in[15];
    const float* bih = (const float*)d_in[16];
    const float* bhh = (const float*)d_in[17];
    float* out = (float*)d_out;

    const int BH  = B_DIM * H_DIM;   // 65536
    float* ws      = (float*)d_ws;
    float* q       = ws;             // [B,A]
    float* logits  = q       + BH;   // [B,S]
    float* weights = logits  + BH;   // [B,S]
    float* attn    = weights + BH;   // [B,C]
    float* x       = attn    + BH;   // [B,IN]
    float* gx      = x       + BH;   // [B,3H]
    float* gh      = gx      + 3 * BH;            // [B,3H]
    _Float16* wk16 = (_Float16*)(gh + 3 * BH);    // [A,C] f16 (2 MB)

    // 0) one-time Wk f32 -> f16 (amortized across all 512 attention WGs)
    cvt_f16_kernel<<<(A_DIM * C_DIM / 4) / 256, 256, 0, stream>>>(Wk, wk16, A_DIM * C_DIM / 4);

    // 1) q = hidden @ Wq.T + bq   (256 wave-tiles -> 32 blocks of 8 waves)
    gemm_xwt_bias<<<32, 256, 0, stream>>>(hidden, Wq, bq, nullptr, nullptr, nullptr,
                                          q, B_DIM, A_DIM, H_DIM, 0);

    // 2) fused attention logits (dominant GEMM, 137 GFLOP)
    const size_t smem = (size_t)STILE * LDH * sizeof(_Float16)
                      + (size_t)A_DIM * sizeof(float)
                      + (size_t)STILE * sizeof(float);          // ~269 KB of 320 KB WGP LDS
    attn_logits_kernel<<<dim3(S_DIM / STILE, B_DIM), 256, smem, stream>>>(
        context, wk16, q, bk, Wl, bl, logits);

    // 3) softmax over S
    softmax_kernel<<<B_DIM, 256, 0, stream>>>(logits, mask, weights);

    // 4) attn = weights @ context  (bandwidth-bound, S split 8-way + f32 atomics)
    zero_kernel<<<BH / 256, 256, 0, stream>>>(attn, BH);
    attn_wsum_kernel<<<dim3(B_DIM, 8), 256, 0, stream>>>(weights, context, attn, S_DIM / 8);

    // 5) x = input @ We.T + be + attn @ Wa.T + ba   (concat-K WMMA GEMM)
    gemm_xwt_bias<<<32, 256, 0, stream>>>(input, We, be, attn, Wa, ba,
                                          x, B_DIM, H_DIM, H_DIM, H_DIM);

    // 6) GRU gate GEMMs: [64,3072] each
    gemm_xwt_bias<<<96, 256, 0, stream>>>(x, Wih, bih, nullptr, nullptr, nullptr,
                                          gx, B_DIM, 3 * H_DIM, H_DIM, 0);
    gemm_xwt_bias<<<96, 256, 0, stream>>>(hidden, Whh, bhh, nullptr, nullptr, nullptr,
                                          gh, B_DIM, 3 * H_DIM, H_DIM, 0);

    // 7) elementwise GRU + emit (hidden1, attn_values)
    gru_kernel<<<BH / 256, 256, 0, stream>>>(gx, gh, hidden, attn, out);

    (void)in_sizes; (void)n_in; (void)out_size; (void)ws_size;
}
